// EvolvingLocalConvBlock_45337674776752
// MI455X (gfx1250) — matrix operand order repaired and checked by reading, not compiled, not run on hardware
//
#include <hip/hip_runtime.h>
#include <hip/hip_bf16.h>
#include <math.h>

// Problem constants (from the reference): B=4, L=2048, D=512, P=128, V=8, K=4
#define BB 4
#define LL 2048
#define DD 512
#define PP 128
#define VV 8
#define KK 4
#define MM (BB*LL)   // 8192 tokens

typedef __attribute__((ext_vector_type(16))) __bf16 v16bf;
typedef __attribute__((ext_vector_type(8)))  float  v8f;

union AFrag { v16bf v; __bf16 e[16]; };
union CFrag { v8f  v; float  e[8]; };
union Pack4 { __bf16 h[4]; uint2 u; };

__device__ __forceinline__ float gelu_exact(float v) {
    return 0.5f * v * (1.0f + erff(v * 0.7071067811865476f));
}
__device__ __forceinline__ float sigmoid_f(float v) {
    return 1.0f / (1.0f + __expf(-v));
}

// ---------------------------------------------------------------------------
// Generic tiled GEMM:  C[M,N] = act(A[M,K] @ W[N,K]^T + bias) (+ resid)
// Block tile 64(M) x 64(N) x 32(K); 8 waves (4 along M x 2 along N); each
// wave owns two 16x16 WMMA tiles sharing one A fragment.
// LDS double-buffered (ping-pong): one barrier per K-step; global loads for
// step s+1 are issued before the WMMAs of step s (software pipelining).
// All staging loads are unconditional float4 with clamped addresses and
// cndmask zero-select (no EXEC-mask branching).
// Fragment layouts used here (A and B share one K order):
//   A 16x32 bf16: lane m(0..15)/half h; e[0..15] = K 16h..16h+15 (contiguous)
//   B 32x16 bf16: lane col n, half h;   e[0..15] = K 16h..16h+15 (contiguous)
//   C 16x16 f32 : lane col n, half h;   VGPR r -> M = r + 8h
// ACT: 0=none 1=gelu 2=sigmoid 3=tanh*pi
// Requirements: M % 64 == 0 (true: M=8192), K % 4 == 0, 16B-aligned pointers.
// ---------------------------------------------------------------------------
template<int ACT, bool RES>
__global__ void __launch_bounds__(256)
gemm_bf16_wmma(const float* __restrict__ A, const float* __restrict__ W,
               const float* __restrict__ bias, const float* __restrict__ resid,
               float* __restrict__ C, int M, int N, int K, int ldc)
{
    __shared__ __bf16 As[2][64][32];   // [buf][m][k]
    __shared__ __bf16 Bs[2][64][32];   // [buf][n][k]

    const int tid  = threadIdx.x;
    const int lane = tid & 31;
    const int wave = tid >> 5;
    const int wm   = wave & 3;            // M subtile 0..3  (rows wm*16..)
    const int wn   = wave >> 2;           // N pair   0..1  (cols wn*32..)
    const int mBase = blockIdx.x * 64;
    const int nBase = blockIdx.y * 64;
    const int lm = lane & 15;
    const int lh = lane >> 4;

    // per-thread staging coordinates: 64 rows x 8 float4 columns
    const int r0  = tid >> 3;             // element 0: rows 0..31
    const int r1  = r0 + 32;              // element 1: rows 32..63
    const int c4  = (tid & 7) * 4;        // k column within tile (0,4,..,28)

    CFrag acc0, acc1;
#pragma unroll
    for (int r = 0; r < 8; ++r) { acc0.e[r] = 0.0f; acc1.e[r] = 0.0f; }

    const int ksteps = (K + 31) / 32;

    auto loadA = [&](int k0, float4& va0, float4& va1) {
        int gk = k0 + c4;
        int kc = (gk < K) ? gk : (K - 4);
        int ra = mBase + r0; if (ra >= M) ra = M - 1;
        int rb = mBase + r1; if (rb >= M) rb = M - 1;
        va0 = *(const float4*)(A + (size_t)ra * K + kc);
        va1 = *(const float4*)(A + (size_t)rb * K + kc);
        if (gk >= K) { va0 = float4{0.f,0.f,0.f,0.f}; va1 = float4{0.f,0.f,0.f,0.f}; }
    };
    auto loadB = [&](int k0, float4& vb0, float4& vb1) {
        int gk = k0 + c4;
        int kc = (gk < K) ? gk : (K - 4);
        int na = nBase + r0;
        int nb = nBase + r1;
        int nca = (na < N) ? na : (N - 1);
        int ncb = (nb < N) ? nb : (N - 1);
        vb0 = *(const float4*)(W + (size_t)nca * K + kc);
        vb1 = *(const float4*)(W + (size_t)ncb * K + kc);
        if (gk >= K || na >= N) vb0 = float4{0.f,0.f,0.f,0.f};
        if (gk >= K || nb >= N) vb1 = float4{0.f,0.f,0.f,0.f};
    };
    auto stage = [&](int buf, const float4& va0, const float4& va1,
                     const float4& vb0, const float4& vb1) {
        Pack4 p;
        p.h[0] = (__bf16)va0.x; p.h[1] = (__bf16)va0.y;
        p.h[2] = (__bf16)va0.z; p.h[3] = (__bf16)va0.w;
        *(uint2*)&As[buf][r0][c4] = p.u;
        p.h[0] = (__bf16)va1.x; p.h[1] = (__bf16)va1.y;
        p.h[2] = (__bf16)va1.z; p.h[3] = (__bf16)va1.w;
        *(uint2*)&As[buf][r1][c4] = p.u;
        p.h[0] = (__bf16)vb0.x; p.h[1] = (__bf16)vb0.y;
        p.h[2] = (__bf16)vb0.z; p.h[3] = (__bf16)vb0.w;
        *(uint2*)&Bs[buf][r0][c4] = p.u;
        p.h[0] = (__bf16)vb1.x; p.h[1] = (__bf16)vb1.y;
        p.h[2] = (__bf16)vb1.z; p.h[3] = (__bf16)vb1.w;
        *(uint2*)&Bs[buf][r1][c4] = p.u;
    };

    float4 va0, va1, vb0, vb1;
    loadA(0, va0, va1);
    loadB(0, vb0, vb1);
    stage(0, va0, va1, vb0, vb1);
    __syncthreads();

    for (int s = 0; s < ksteps; ++s) {
        const int buf = s & 1;
        const bool more = (s + 1) < ksteps;
        if (more) {
            // L2 prefetch two steps ahead (gfx1250 global_prefetch_b8)
            if ((s + 2) * 32 < K)
                __builtin_prefetch(A + (size_t)(mBase + (tid & 63)) * K + (s + 2) * 32, 0, 1);
            loadA((s + 1) * 32, va0, va1);
            loadB((s + 1) * 32, vb0, vb1);
        }

        // fragments from current buffer
        AFrag a, b0, b1;
        const __bf16* ap  = &As[buf][wm * 16 + lm][0];
        *(uint4*)&a.e[0] = *(const uint4*)(ap + 16 * lh);
        *(uint4*)&a.e[8] = *(const uint4*)(ap + 16 * lh + 8);
        const __bf16* bp0 = &Bs[buf][wn * 32 + lm][0];
        const __bf16* bp1 = &Bs[buf][wn * 32 + 16 + lm][0];
        *(uint4*)&b0.e[0] = *(const uint4*)(bp0 + 16 * lh);
        *(uint4*)&b0.e[8] = *(const uint4*)(bp0 + 16 * lh + 8);
        *(uint4*)&b1.e[0] = *(const uint4*)(bp1 + 16 * lh);
        *(uint4*)&b1.e[8] = *(const uint4*)(bp1 + 16 * lh + 8);

        acc0.v = __builtin_amdgcn_wmma_f32_16x16x32_bf16(
            false, a.v, false, b0.v, (short)0, acc0.v, false, false);
        acc1.v = __builtin_amdgcn_wmma_f32_16x16x32_bf16(
            false, a.v, false, b1.v, (short)0, acc1.v, false, false);

        if (more) stage(buf ^ 1, va0, va1, vb0, vb1);
        __syncthreads();
    }

    // writeback both 16x16 tiles
#pragma unroll
    for (int t = 0; t < 2; ++t) {
        const CFrag& acc = t ? acc1 : acc0;
        int col = nBase + wn * 32 + t * 16 + lm;
        if (col < N) {
            float bv = bias ? bias[col] : 0.0f;
#pragma unroll
            for (int r = 0; r < 8; ++r) {
                int row = mBase + wm * 16 + r + 8 * lh;
                float v = acc.e[r] + bv;
                if (ACT == 1) v = gelu_exact(v);
                else if (ACT == 2) v = sigmoid_f(v);
                else if (ACT == 3) v = tanhf(v) * 3.14159265358979323846f;
                if (RES) v += resid[(size_t)row * N + col];
                C[(size_t)row * ldc + col] = v;
            }
        }
    }
}

// ---------------------------------------------------------------------------
// Depthwise causal conv (K=4) with sigmoid gate
// ---------------------------------------------------------------------------
__global__ void dwconv_gate_kernel(const float* __restrict__ x,
                                   const float* __restrict__ lcw, const float* __restrict__ lcb,
                                   const float* __restrict__ cgw, const float* __restrict__ cgb,
                                   float* __restrict__ out)
{
    int idx = blockIdx.x * blockDim.x + threadIdx.x;   // over M*D
    if (idx >= MM * DD) return;
    int d = idx % DD;
    int t = idx / DD;
    int l = t % LL;
    float co = lcb[d], cg = cgb[d];
#pragma unroll
    for (int k = 0; k < KK; ++k) {
        int ll = l + k - (KK - 1);
        if (ll >= 0) {
            float xv = x[(size_t)(t + k - (KK - 1)) * DD + d];
            co += lcw[d * KK + k] * xv;
            cg += cgw[d * KK + k] * xv;
        }
    }
    out[idx] = co * sigmoid_f(cg);
}

// ---------------------------------------------------------------------------
// store_gate = sigmoid(x @ sg_w^T + sg_b)  (N=1: one wave per token)
// ---------------------------------------------------------------------------
__global__ void rowdot_sigmoid_kernel(const float* __restrict__ x,
                                      const float* __restrict__ w,
                                      const float* __restrict__ b,
                                      float* __restrict__ out)
{
    int tok  = (blockIdx.x * blockDim.x + threadIdx.x) >> 5;
    int lane = threadIdx.x & 31;
    if (tok >= MM) return;
    float s = 0.0f;
    for (int k = lane; k < DD; k += 32) s += x[(size_t)tok * DD + k] * w[k];
#pragma unroll
    for (int off = 16; off; off >>= 1) s += __shfl_xor(s, off, 32);
    if (lane == 0) out[tok] = sigmoid_f(s + b[0]);
}

// ---------------------------------------------------------------------------
// Fused per-(b,d) channel scan over L: phi cumsum, magnitude cumsums,
// pos_ret, x*cos/x*sin (straight into `combined`), context_avg + sk0 concat.
// ---------------------------------------------------------------------------
__global__ void channel_scan_kernel(const float* __restrict__ x,
                                    const float* __restrict__ omega_pre,
                                    const float* __restrict__ omega_scale,
                                    const float* __restrict__ phi_base,
                                    const float* __restrict__ v1,
                                    const float* __restrict__ magn,
                                    const float* __restrict__ mag_scale,
                                    const float* __restrict__ qoff,
                                    float* __restrict__ pos_ret,
                                    float* __restrict__ combined,
                                    float* __restrict__ catbuf)
{
    int idx = blockIdx.x * blockDim.x + threadIdx.x;
    if (idx >= BB * DD) return;
    int b = idx / DD, d = idx % DD;
    float os  = fabsf(omega_scale[d]);
    float msc = fabsf(mag_scale[0]);
    float so = 0.f, sm = 0.f, sc = 0.f, ss = 0.f, sx = 0.f;
    const float invSqrtD = 0.044194173824159216f;   // 1/sqrt(512)
    for (int l = 0; l < LL; ++l) {
        size_t t = (size_t)b * LL + l;
        size_t i = t * DD + d;
        so += omega_pre[i] * os;
        float phi = phi_base[i] + so;
        float c, s;  __sincosf(phi, &s, &c);
        float mg = magn[i] * msc;
        float wv = mg * v1[i];
        sm += mg; sc += wv * c; ss += wv * s;
        float inv = rsqrtf(sm + 1e-8f);
        float phiq = phi + qoff[i];
        float cq, sq; __sincosf(phiq, &sq, &cq);
        pos_ret[i] = (sc * cq + ss * sq) * inv * invSqrtD;
        float xv = x[i];
        combined[t * (5 * DD) + 3 * DD + d] = xv * c;
        combined[t * (5 * DD) + 4 * DD + d] = xv * s;
        sx += xv;
        catbuf[t * (2 * DD) + d]      = xv;
        catbuf[t * (2 * DD) + DD + d] = sx / (float)(l + 1);
    }
}

// ---------------------------------------------------------------------------
// Phase-keyed KV memory scan: block per batch, thread = p (128), V=8 state
// in registers; P-reduction via LDS ds_add_f32 atomics.
// ---------------------------------------------------------------------------
__global__ void __launch_bounds__(PP)
kv_scan_kernel(const float* __restrict__ sphase, const float* __restrict__ qphase,
               const float* __restrict__ values, const float* __restrict__ sgate,
               float* __restrict__ kvret)
{
    int b = blockIdx.x;
    int p = threadIdx.x;
    __shared__ float acc[VV];
    float mc[VV], ms[VV];
#pragma unroll
    for (int v = 0; v < VV; ++v) { mc[v] = 0.f; ms[v] = 0.f; }
    float gs = 0.f;
    const float invSqrtP = 0.08838834764831845f;    // 1/sqrt(128)
    for (int l = 0; l < LL; ++l) {
        int t = b * LL + l;
        float g = sgate[t];
        gs += g;
        float gn = sqrtf(fmaxf(gs, 1.0f));
        float sp = sphase[(size_t)t * PP + p];
        float cs, sn; __sincosf(sp, &sn, &cs);
#pragma unroll
        for (int v = 0; v < VV; ++v) {
            float gv = values[(size_t)t * VV + v] * g;
            mc[v] += cs * gv;
            ms[v] += sn * gv;
        }
        float qp = qphase[(size_t)t * PP + p];
        float cq, sq; __sincosf(qp, &sq, &cq);
        if (p < VV) acc[p] = 0.f;
        __syncthreads();
#pragma unroll
        for (int v = 0; v < VV; ++v) atomicAdd(&acc[v], cq * mc[v] + sq * ms[v]);
        __syncthreads();
        if (p < VV) kvret[(size_t)t * VV + p] = acc[p] * invSqrtP / gn;
        __syncthreads();
    }
}

// ---------------------------------------------------------------------------
// In-place LayerNorm over N=2560 per token (block per token).
// Row is staged into LDS with gfx1250 async LDS loads
// (global_load_async_to_lds_b128, ASYNCcnt) so the mean/var passes read LDS,
// halving global read traffic for this kernel.
// ---------------------------------------------------------------------------
__global__ void __launch_bounds__(256)
layernorm_kernel(float* __restrict__ h, const float* __restrict__ g,
                 const float* __restrict__ bta)
{
    constexpr int N = 5 * DD;            // 2560
    __shared__ float rowsh[N];           // 10 KB
    __shared__ float red[256];
    float* row = h + (size_t)blockIdx.x * N;

    unsigned ldsbase = (unsigned)(uintptr_t)&rowsh[0];   // addr[31:0] = LDS offset
    for (int i = threadIdx.x; i < N / 4; i += 256) {
        unsigned lds = ldsbase + (unsigned)i * 16u;
        const float* gp = row + i * 4;
        asm volatile("global_load_async_to_lds_b128 %0, %1, off"
                     :: "v"(lds), "v"(gp) : "memory");
    }
    asm volatile("s_wait_asynccnt 0x0" ::: "memory");
    __syncthreads();

    float s = 0.f;
    for (int i = threadIdx.x; i < N; i += 256) s += rowsh[i];
    red[threadIdx.x] = s; __syncthreads();
    for (int off = 128; off; off >>= 1) {
        if (threadIdx.x < off) red[threadIdx.x] += red[threadIdx.x + off];
        __syncthreads();
    }
    float mean = red[0] / N; __syncthreads();
    float vs = 0.f;
    for (int i = threadIdx.x; i < N; i += 256) { float dv = rowsh[i] - mean; vs += dv * dv; }
    red[threadIdx.x] = vs; __syncthreads();
    for (int off = 128; off; off >>= 1) {
        if (threadIdx.x < off) red[threadIdx.x] += red[threadIdx.x + off];
        __syncthreads();
    }
    float rstd = rsqrtf(red[0] / N + 1e-5f);
    for (int i = threadIdx.x; i < N; i += 256)
        row[i] = (rowsh[i] - mean) * rstd * g[i] + bta[i];
}

// ---------------------------------------------------------------------------
extern "C" void kernel_launch(void* const* d_in, const int* in_sizes, int n_in,
                              void* d_out, int out_size, void* d_ws, size_t ws_size,
                              hipStream_t stream)
{
    (void)in_sizes; (void)n_in; (void)out_size; (void)ws_size;
    const float* x           = (const float*)d_in[0];
    const float* lc_w        = (const float*)d_in[1];
    const float* lc_b        = (const float*)d_in[2];
    const float* cg_w        = (const float*)d_in[3];
    const float* cg_b        = (const float*)d_in[4];
    const float* cp_w        = (const float*)d_in[5];
    const float* cp_b        = (const float*)d_in[6];
    const float* tw_w        = (const float*)d_in[7];
    const float* tw_b        = (const float*)d_in[8];
    const float* omega_scale = (const float*)d_in[9];
    const float* pi0_w       = (const float*)d_in[10];
    const float* pi0_b       = (const float*)d_in[11];
    const float* pi2_w       = (const float*)d_in[12];
    const float* pi2_b       = (const float*)d_in[13];
    const float* m1v_w       = (const float*)d_in[14];
    const float* m1v_b       = (const float*)d_in[15];
    const float* m1o_w       = (const float*)d_in[16];
    const float* m1o_b       = (const float*)d_in[17];
    const float* mag_w       = (const float*)d_in[18];
    const float* mag_b       = (const float*)d_in[19];
    const float* mag_scale   = (const float*)d_in[20];
    const float* qo_w        = (const float*)d_in[21];
    const float* qo_b        = (const float*)d_in[22];
    const float* ke_w        = (const float*)d_in[23];
    const float* ke_b        = (const float*)d_in[24];
    const float* ve_w        = (const float*)d_in[25];
    const float* ve_b        = (const float*)d_in[26];
    const float* sk0_w       = (const float*)d_in[27];
    const float* sk0_b       = (const float*)d_in[28];
    const float* sk2_w       = (const float*)d_in[29];
    const float* sk2_b       = (const float*)d_in[30];
    const float* sg_w        = (const float*)d_in[31];
    const float* sg_b        = (const float*)d_in[32];
    const float* kvo_w       = (const float*)d_in[33];
    const float* kvo_b       = (const float*)d_in[34];
    const float* ln_g        = (const float*)d_in[35];
    const float* ln_b        = (const float*)d_in[36];
    const float* o1_w        = (const float*)d_in[37];
    const float* o1_b        = (const float*)d_in[38];
    const float* o2_w        = (const float*)d_in[39];
    const float* o2_b        = (const float*)d_in[40];
    float* out = (float*)d_out;

    float* ws = (float*)d_ws;
    size_t off = 0;
    auto alloc = [&](size_t n) { float* p = ws + off; off += n; return p; };
    const size_t MD = (size_t)MM * DD;
    float* omega_pre = alloc(MD);
    float* t0        = alloc(MD);        // gelu(pi0(x))
    float* phi_base  = alloc(MD);        // pi2(t0)
    float* v1        = alloc(MD);
    float* magn      = alloc(MD);        // sigmoid(mag(x)); |mag_scale| in scan
    float* qoff      = alloc(MD);
    float* pos_ret   = alloc(MD);
    float* convg     = alloc(MD);        // gated depthwise-conv output
    float* t1        = alloc(MD);        // gelu(sk0(cat))
    float* catbuf    = alloc(2 * MD);    // [x, context_avg]
    float* h1        = alloc(2 * MD);    // gelu(o1(...))
    float* combined  = alloc(5 * MD);    // [conv_out, pos_out, kv_out, x*cos, x*sin]
    float* sphase    = alloc((size_t)MM * PP);
    float* qphase    = alloc((size_t)MM * PP);
    float* values    = alloc((size_t)MM * VV);
    float* sgate     = alloc((size_t)MM);
    float* kvret     = alloc((size_t)MM * VV);

    dim3 blk(256);
    auto gg = [](int M, int N) { return dim3((M + 63) / 64, (N + 63) / 64); };

    // --- token-parallel GEMMs on x (pi2 runs on t0 from pi0) ---
    gemm_bf16_wmma<0,false><<<gg(MM,DD), blk, 0, stream>>>(x, tw_w,  tw_b,  nullptr, omega_pre, MM, DD, DD, DD);
    gemm_bf16_wmma<1,false><<<gg(MM,DD), blk, 0, stream>>>(x, pi0_w, pi0_b, nullptr, t0,        MM, DD, DD, DD);
    gemm_bf16_wmma<0,false><<<gg(MM,DD), blk, 0, stream>>>(t0, pi2_w, pi2_b, nullptr, phi_base, MM, DD, DD, DD);
    gemm_bf16_wmma<0,false><<<gg(MM,DD), blk, 0, stream>>>(x, m1v_w, m1v_b, nullptr, v1,        MM, DD, DD, DD);
    gemm_bf16_wmma<2,false><<<gg(MM,DD), blk, 0, stream>>>(x, mag_w, mag_b, nullptr, magn,      MM, DD, DD, DD);
    gemm_bf16_wmma<0,false><<<gg(MM,DD), blk, 0, stream>>>(x, qo_w,  qo_b,  nullptr, qoff,      MM, DD, DD, DD);
    gemm_bf16_wmma<3,false><<<gg(MM,PP), blk, 0, stream>>>(x, ke_w,  ke_b,  nullptr, qphase,    MM, PP, DD, PP);
    gemm_bf16_wmma<0,false><<<gg(MM,VV), blk, 0, stream>>>(x, ve_w,  ve_b,  nullptr, values,    MM, VV, DD, VV);
    rowdot_sigmoid_kernel<<<dim3((MM * 32 + 255) / 256), blk, 0, stream>>>(x, sg_w, sg_b, sgate);

    // --- conv branch ---
    dwconv_gate_kernel<<<dim3((MM * DD + 255) / 256), blk, 0, stream>>>(x, lc_w, lc_b, cg_w, cg_b, convg);
    gemm_bf16_wmma<0,false><<<gg(MM,DD), blk, 0, stream>>>(convg, cp_w, cp_b, nullptr, combined + 0, MM, DD, DD, 5 * DD);

    // --- fused channel scans (phi, mem1, pos_ret, x*cos/sin, context_avg) ---
    channel_scan_kernel<<<dim3((BB * DD + 255) / 256), blk, 0, stream>>>(
        x, omega_pre, omega_scale, phi_base, v1, magn, mag_scale, qoff,
        pos_ret, combined, catbuf);
    gemm_bf16_wmma<0,false><<<gg(MM,DD), blk, 0, stream>>>(pos_ret, m1o_w, m1o_b, nullptr, combined + DD, MM, DD, DD, 5 * DD);

    // --- storage phase path + KV scan ---
    gemm_bf16_wmma<1,false><<<gg(MM,DD), blk, 0, stream>>>(catbuf, sk0_w, sk0_b, nullptr, t1, MM, DD, 2 * DD, DD);
    gemm_bf16_wmma<3,false><<<gg(MM,PP), blk, 0, stream>>>(t1, sk2_w, sk2_b, nullptr, sphase, MM, PP, DD, PP);
    kv_scan_kernel<<<dim3(BB), dim3(PP), 0, stream>>>(sphase, qphase, values, sgate, kvret);
    gemm_bf16_wmma<0,false><<<gg(MM,DD), blk, 0, stream>>>(kvret, kvo_w, kvo_b, nullptr, combined + 2 * DD, MM, DD, VV, 5 * DD);

    // --- combine: LN -> o1(gelu) -> o2 (+ residual) ---
    layernorm_kernel<<<dim3(MM), blk, 0, stream>>>(combined, ln_g, ln_b);
    gemm_bf16_wmma<1,false><<<gg(MM, 2 * DD), blk, 0, stream>>>(combined, o1_w, o1_b, nullptr, h1, MM, 2 * DD, 5 * DD, 2 * DD);
    gemm_bf16_wmma<0,true ><<<gg(MM, DD),     blk, 0, stream>>>(h1, o2_w, o2_b, x, out, MM, DD, 2 * DD, DD);
}
